// JointAttention_66374424593004
// MI455X (gfx1250) — compile-verified, matrix operand order checked by reading, not run on hardware
//
#include <hip/hip_runtime.h>
#include <hip/hip_bf16.h>

// Problem constants (match reference)
#define DIMM   2048
#define NH     16
#define NKV    8
#define HD     128
#define BSZ    2
#define SEQ    2048
#define NTOK   (BSZ * SEQ)               // 4096 tokens
#define QKVD   ((NH + 2 * NKV) * HD)     // 4096
#define EPSF   1e-5f

typedef __attribute__((ext_vector_type(16))) __bf16       v16bf;
typedef __attribute__((ext_vector_type(8)))  float        v8f;
typedef __attribute__((ext_vector_type(4)))  unsigned int u32x4;
typedef __attribute__((ext_vector_type(2)))  unsigned int u32x2;

#if defined(__has_builtin)
#if __has_builtin(__builtin_amdgcn_tensor_load_to_lds)
#define HAS_TDM 1
#endif
#endif
#ifndef HAS_TDM
#define HAS_TDM 0
#endif

// ---------- helpers ----------

__device__ inline unsigned short f2bf(float f) {
    unsigned int u = __float_as_uint(f);
    unsigned int r = (u + 0x7FFFu + ((u >> 16) & 1u)) >> 16;   // RNE
    return (unsigned short)r;
}

// Load a 16x32 bf16 fragment (A-layout, also used for B = W[n][k] row-major):
// lane L -> row/col (L&15), half (L>>4). Elements 0..7 = K half*8..+7,
// elements 8..15 = K 16+half*8..+7 (CDNA5 ISA 7.12.2, 16-bit A 16x32 layout).
__device__ inline v16bf load_frag_k32(const unsigned short* base, int stride, int lane) {
    int r  = lane & 15;
    int hf = lane >> 4;
    const unsigned short* p = base + (size_t)r * stride + hf * 8;
    u32x4 lo = *(const u32x4*)p;
    u32x4 hi = *(const u32x4*)(p + 16);
    union { u32x4 q[2]; v16bf f; } u;
    u.q[0] = lo; u.q[1] = hi;
    return u.f;
}

__device__ inline float red_max16(float v) {
#pragma unroll
    for (int m = 1; m < 16; m <<= 1) v = fmaxf(v, __shfl_xor(v, m, 32));
    return v;
}
__device__ inline float red_sum16(float v) {
#pragma unroll
    for (int m = 1; m < 16; m <<= 1) v += __shfl_xor(v, m, 32);
    return v;
}
__device__ inline float red_sum32(float v) {
#pragma unroll
    for (int m = 1; m < 32; m <<= 1) v += __shfl_xor(v, m, 32);
    return v;
}

#if HAS_TDM
// ---------- Tensor Data Mover: 2D tile global -> LDS ----------
typedef __attribute__((ext_vector_type(4))) unsigned int tdmv4u;
typedef __attribute__((ext_vector_type(8))) int          tdmv8i;
typedef __attribute__((ext_vector_type(4))) int          tdmv4i;

// D# per CDNA5 ISA ch.8: group0 = {flags, lds_addr, global_addr, type=2},
// group1 = {data_size=2B, tensor dims, tile dims 32 x rows, dim0 stride}.
// Groups 2/3 zero (2D tensor, no iteration/gather).
// 6-arg builtin form (clang-23 / therock-10.0 lane): the 5th operand is an
// extra int32x8 group, zero-filled here.
__device__ inline void tdm_load_2d(unsigned lds_off, const void* gaddr,
                                   unsigned tile_rows,
                                   unsigned long long row_stride_elems) {
    unsigned long long ga = (unsigned long long)(size_t)gaddr;
    const unsigned long long td0 = 0x100000ull;   // tensor_dim0 (large: no OOB clip)
    const unsigned long long td1 = 0x100000ull;   // tensor_dim1
    tdmv4u g0;
    g0[0] = 1u;                                        // count=1, user descriptor
    g0[1] = lds_off;                                   // lds_addr (bytes)
    g0[2] = (unsigned)(ga & 0xFFFFFFFFull);            // global_addr[31:0]
    g0[3] = (unsigned)((ga >> 32) & 0x1FFFFFFull)      // global_addr[56:32]
          | (2u << 30);                                // type = 2 ("image")
    tdmv8i g1;
    g1[0] = (int)(1u << 16);                                           // data_size=1 -> 2 bytes
    g1[1] = (int)((td0 & 0xFFFFull) << 16);                            // tensor_dim0[15:0]
    g1[2] = (int)(((td0 >> 16) & 0xFFFFull) | ((td1 & 0xFFFFull) << 16));
    g1[3] = (int)(((td1 >> 16) & 0xFFFFull) | (32u << 16));            // tile_dim0 = 32
    g1[4] = (int)(tile_rows & 0xFFFFu);                                // tile_dim1, tile_dim2=0
    g1[5] = (int)(row_stride_elems & 0xFFFFFFFFull);                   // dim0_stride[31:0]
    g1[6] = (int)((row_stride_elems >> 32) & 0xFFFFull);               // dim0_stride[47:32]
    g1[7] = 0;
    tdmv4i z4 = {0, 0, 0, 0};
    tdmv8i z8 = {0, 0, 0, 0, 0, 0, 0, 0};
    __builtin_amdgcn_tensor_load_to_lds(g0, g1, z4, z4, z8, 0);
}

__device__ inline unsigned lds_off_of(const void* p) {
    return (unsigned)(size_t)p;   // generic LDS pointer: low 32 bits = LDS offset
}
#endif

// ---------- kernels ----------

__global__ void __launch_bounds__(256) cvt_bf16_kernel(const float* __restrict__ in,
                                                       unsigned short* __restrict__ out,
                                                       long long n) {
    long long i = ((long long)blockIdx.x * 256 + threadIdx.x) * 4;
    if (i + 3 >= n) { if (i < n) { for (long long j = i; j < n; ++j) out[j] = f2bf(in[j]); } return; }
    float4 v = *(const float4*)(in + i);
    unsigned int lo = (unsigned int)f2bf(v.x) | ((unsigned int)f2bf(v.y) << 16);
    unsigned int hi = (unsigned int)f2bf(v.z) | ((unsigned int)f2bf(v.w) << 16);
    u32x2 pv = {lo, hi};
    *(u32x2*)(out + i) = pv;
}

// C[M][N] (f32) = A[M][K] (bf16 row-major) * B[N][K]^T (bf16 row-major).
// Block = 4 waves = 64x128 C tile; per-wave 16x128 (8 accumulators).
// A (64x32) and B (128x32) K-slices are staged in LDS. With TDM available,
// wave 0 double-buffers the tiles via tensor_load_to_lds and s_wait_tensorcnt,
// overlapping the global->LDS DMA of slice k+32 with WMMA on slice k.
// All 8 B fragments are preloaded before the WMMA chain so ds_loads pipeline
// instead of draining DScnt before every WMMA.
__global__ void __launch_bounds__(128) wmma_gemm_nt(const unsigned short* __restrict__ A,
                                                    const unsigned short* __restrict__ Bw,
                                                    float* __restrict__ C,
                                                    int M, int N, int K) {
    __shared__ __align__(16) unsigned short Atile[2][64 * 32];
    __shared__ __align__(16) unsigned short Btile[2][128 * 32];

    const int lane  = threadIdx.x & 31;
    const int wave  = threadIdx.x >> 5;
    const int tid   = threadIdx.x;
    const int ntile = blockIdx.x;               // 128 cols
    const int mblk  = blockIdx.y;               // 64 rows
    const unsigned short* abase = A  + (size_t)mblk  * 64  * K;
    const unsigned short* bbase = Bw + (size_t)ntile * 128 * K;

    v8f acc[8] = {};

#if HAS_TDM
    if (wave == 0) {
        tdm_load_2d(lds_off_of(&Atile[0][0]), abase, 64,  (unsigned long long)K);
        tdm_load_2d(lds_off_of(&Btile[0][0]), bbase, 128, (unsigned long long)K);
    }
    for (int k = 0; k < K; k += 32) {
        const int cur = (k >> 5) & 1;
        if (wave == 0) {
            if (k + 32 < K) {
                tdm_load_2d(lds_off_of(&Atile[cur ^ 1][0]), abase + k + 32, 64,
                            (unsigned long long)K);
                tdm_load_2d(lds_off_of(&Btile[cur ^ 1][0]), bbase + k + 32, 128,
                            (unsigned long long)K);
                __builtin_amdgcn_s_wait_tensorcnt(2);   // current slice's 2 DMAs done
            } else {
                __builtin_amdgcn_s_wait_tensorcnt(0);
            }
        }
        __syncthreads();
        v16bf af = load_frag_k32(&Atile[cur][wave * 16 * 32], 32, lane);
        v16bf bf[8];
#pragma unroll
        for (int j = 0; j < 8; ++j)
            bf[j] = load_frag_k32(&Btile[cur][j * 16 * 32], 32, lane);
#pragma unroll
        for (int j = 0; j < 8; ++j)
            acc[j] = __builtin_amdgcn_wmma_f32_16x16x32_bf16(
                false, af, false, bf[j], (short)0, acc[j], false, false);
        __syncthreads();   // all waves done reading before buffer is re-filled
    }
#else
    for (int k = 0; k < K; k += 32) {
        __syncthreads();
        // cooperative staging: A 64x32 (16 elems/thread), B 128x32 (32 elems/thread)
#pragma unroll
        for (int c = 0; c < 2; ++c) {
            int idx = c * 128 + tid;            // 256 chunks of 8 elems
            int row = idx >> 2, dp = (idx & 3) * 8;
            *(u32x4*)(&Atile[0][row * 32 + dp]) =
                *(const u32x4*)(abase + (size_t)row * K + k + dp);
        }
#pragma unroll
        for (int c = 0; c < 4; ++c) {
            int idx = c * 128 + tid;            // 512 chunks of 8 elems
            int row = idx >> 2, dp = (idx & 3) * 8;
            *(u32x4*)(&Btile[0][row * 32 + dp]) =
                *(const u32x4*)(bbase + (size_t)row * K + k + dp);
        }
        __syncthreads();
        v16bf af = load_frag_k32(&Atile[0][wave * 16 * 32], 32, lane);
        v16bf bf[8];
#pragma unroll
        for (int j = 0; j < 8; ++j)
            bf[j] = load_frag_k32(&Btile[0][j * 16 * 32], 32, lane);
#pragma unroll
        for (int j = 0; j < 8; ++j)
            acc[j] = __builtin_amdgcn_wmma_f32_16x16x32_bf16(
                false, af, false, bf[j], (short)0, acc[j], false, false);
    }
#endif

    const int half = lane >> 4, col0 = lane & 15;
#pragma unroll
    for (int j = 0; j < 8; ++j) {
#pragma unroll
        for (int i = 0; i < 8; ++i) {
            int row = mblk * 64 + wave * 16 + half * 8 + i;
            int col = ntile * 128 + j * 16 + col0;
            C[(size_t)row * N + col] = acc[j][i];
        }
    }
}

// Per (token, unit): unit 0..15 = Q head (RMSNorm+RoPE), 16..23 = K head
// (RMSNorm+RoPE), 24..31 = V head (bf16 passthrough). One wave per unit.
__global__ void __launch_bounds__(128) rms_rope_scatter(const float* __restrict__ qkv,
                                                        const float* __restrict__ fcos,
                                                        const float* __restrict__ fsin,
                                                        const float* __restrict__ qw,
                                                        const float* __restrict__ kw,
                                                        unsigned short* __restrict__ Qb,
                                                        unsigned short* __restrict__ Kb,
                                                        unsigned short* __restrict__ Vb) {
    const int t    = blockIdx.x;
    const int wave = threadIdx.x >> 5;
    const int u    = blockIdx.y * 4 + wave;      // 0..31
    const int lane = threadIdx.x & 31;
    const int b = t / SEQ, s = t % SEQ;
    const int d0 = lane * 4;

    const float* base = qkv + (size_t)t * QKVD;
    const float* src;
    if (u < NH)            src = base + u * HD;
    else if (u < NH + NKV) src = base + NH * HD + (u - NH) * HD;
    else                   src = base + (NH + NKV) * HD + (u - NH - NKV) * HD;

    float x0 = src[d0 + 0], x1 = src[d0 + 1], x2 = src[d0 + 2], x3 = src[d0 + 3];

    if (u >= NH + NKV) {  // V: no norm / no rope
        unsigned short* dst = Vb + ((size_t)(b * NKV + (u - NH - NKV)) * SEQ + s) * HD + d0;
        dst[0] = f2bf(x0); dst[1] = f2bf(x1); dst[2] = f2bf(x2); dst[3] = f2bf(x3);
        return;
    }

    float ss = x0 * x0 + x1 * x1 + x2 * x2 + x3 * x3;
    ss = red_sum32(ss);
    float inv = rsqrtf(ss * (1.0f / (float)HD) + EPSF);
    const float* w = (u < NH) ? qw : kw;
    x0 *= inv * w[d0 + 0]; x1 *= inv * w[d0 + 1];
    x2 *= inv * w[d0 + 2]; x3 *= inv * w[d0 + 3];

    // RoPE: lane handles pairs p = 2*lane, 2*lane+1
    const int p = lane * 2;
    float c0 = fcos[(size_t)s * 64 + p],     s0 = fsin[(size_t)s * 64 + p];
    float c1 = fcos[(size_t)s * 64 + p + 1], s1 = fsin[(size_t)s * 64 + p + 1];
    float y0 = x0 * c0 - x1 * s0, y1 = x0 * s0 + x1 * c0;
    float y2 = x2 * c1 - x3 * s1, y3 = x2 * s1 + x3 * c1;

    unsigned short* dst = (u < NH)
        ? Qb + ((size_t)(b * NH + u) * SEQ + s) * HD + d0
        : Kb + ((size_t)(b * NKV + (u - NH)) * SEQ + s) * HD + d0;
    dst[0] = f2bf(y0); dst[1] = f2bf(y1); dst[2] = f2bf(y2); dst[3] = f2bf(y3);
}

// Flash attention: block = 4 waves, each wave owns 16 q rows (block = 64 rows).
// K tile (32x128) staged row-major in LDS; V tile staged TRANSPOSED (128x32)
// so the P*V B-fragments are contiguous reads. P goes through LDS to convert
// C-layout scores into an A-layout fragment. K/V fragments are preloaded
// before their WMMA chains to pipeline the ds_loads.
__global__ void __launch_bounds__(128) flash_attn(const unsigned short* __restrict__ Qb,
                                                  const unsigned short* __restrict__ Kb,
                                                  const unsigned short* __restrict__ Vb,
                                                  const unsigned char* __restrict__ xmask,
                                                  unsigned short* __restrict__ Ob) {
    __shared__ __align__(16) unsigned short Kt[32 * HD];   // [key][d]
    __shared__ __align__(16) unsigned short Vt[HD * 32];   // [d][key] (transposed)
    __shared__ __align__(16) unsigned short Pb[4 * 16 * 32];

    const int tid  = threadIdx.x;
    const int lane = tid & 31;
    const int wave = tid >> 5;
    const int qt = blockIdx.x, h = blockIdx.y, b = blockIdx.z;
    const int kvh = h >> 1;                      // n_rep = NH/NKV = 2
    const int q0 = qt * 64 + wave * 16;
    const int half = lane >> 4, nn = lane & 15;

    const unsigned short* qbase = Qb + ((size_t)(b * NH + h) * SEQ + q0) * HD;
    const unsigned short* kbase = Kb + ((size_t)(b * NKV + kvh) * SEQ) * HD;
    const unsigned short* vbase = Vb + ((size_t)(b * NKV + kvh) * SEQ) * HD;

    v16bf qf[4];
#pragma unroll
    for (int kk = 0; kk < 4; ++kk) qf[kk] = load_frag_k32(qbase + kk * 32, HD, lane);

    v8f acc[8] = {};
    float mrow[8], lrow[8];
#pragma unroll
    for (int i = 0; i < 8; ++i) { mrow[i] = -1e30f; lrow[i] = 0.0f; }

    const float scale = 0.08838834764831845f;    // 1/sqrt(128)

    for (int kc = 0; kc < SEQ; kc += 32) {
        __syncthreads();
        // stage K tile [key][d] (coalesced 16B chunks)
#pragma unroll
        for (int c = 0; c < 4; ++c) {
            int idx = c * 128 + tid;
            int key = idx >> 4, dp = (idx & 15) * 8;
            *(u32x4*)(&Kt[key * HD + dp]) =
                *(const u32x4*)(kbase + (size_t)(kc + key) * HD + dp);
        }
        // stage V tile transposed [d][key]
#pragma unroll
        for (int c = 0; c < 4; ++c) {
            int idx = c * 128 + tid;
            int key = idx >> 4, dp = (idx & 15) * 8;
            u32x4 v = *(const u32x4*)(vbase + (size_t)(kc + key) * HD + dp);
            const unsigned short* pv = (const unsigned short*)&v;
#pragma unroll
            for (int e = 0; e < 8; ++e) Vt[(dp + e) * 32 + key] = pv[e];
        }
        if (kc + 32 < SEQ)
            __builtin_prefetch(kbase + (size_t)(kc + 32) * HD + tid * 32, 0, 3);
        __syncthreads();

        // scores: two 16x16 tiles over this 32-key chunk
        v8f st[2];
#pragma unroll
        for (int g = 0; g < 2; ++g) {
            v16bf kfr[4];
#pragma unroll
            for (int kk = 0; kk < 4; ++kk)
                kfr[kk] = load_frag_k32(&Kt[(g * 16) * HD + kk * 32], HD, lane);
            v8f s = {};
#pragma unroll
            for (int kk = 0; kk < 4; ++kk)
                s = __builtin_amdgcn_wmma_f32_16x16x32_bf16(
                    false, qf[kk], false, kfr[kk], (short)0, s, false, false);
            int key = kc + g * 16 + nn;
            float madd = xmask[(size_t)b * SEQ + key] ? 0.0f : -1e30f;
#pragma unroll
            for (int i = 0; i < 8; ++i) s[i] = s[i] * scale + madd;
            st[g] = s;
        }

        // online softmax + P write (bf16) to LDS
#pragma unroll
        for (int i = 0; i < 8; ++i) {
            float cm = red_max16(fmaxf(st[0][i], st[1][i]));
            float mnew  = fmaxf(mrow[i], cm);
            float alpha = __expf(mrow[i] - mnew);
            float p0 = __expf(st[0][i] - mnew);
            float p1 = __expf(st[1][i] - mnew);
            float rs = red_sum16(p0 + p1);
            lrow[i] = lrow[i] * alpha + rs;
            mrow[i] = mnew;
#pragma unroll
            for (int j = 0; j < 8; ++j) acc[j][i] *= alpha;
            Pb[wave * 512 + (i + 8 * half) * 32 + nn]      = f2bf(p0);
            Pb[wave * 512 + (i + 8 * half) * 32 + 16 + nn] = f2bf(p1);
        }
        __syncthreads();

        // P (16x32) * V (32x128) -> accumulate
        v16bf pf = load_frag_k32(&Pb[wave * 512], 32, lane);
        v16bf vfr[8];
#pragma unroll
        for (int j = 0; j < 8; ++j)
            vfr[j] = load_frag_k32(&Vt[(j * 16) * 32], 32, lane);
#pragma unroll
        for (int j = 0; j < 8; ++j)
            acc[j] = __builtin_amdgcn_wmma_f32_16x16x32_bf16(
                false, pf, false, vfr[j], (short)0, acc[j], false, false);
    }

    // epilogue: normalize and write bf16 [token][h*128+d] for the out-proj GEMM
#pragma unroll
    for (int j = 0; j < 8; ++j) {
#pragma unroll
        for (int i = 0; i < 8; ++i) {
            int row = q0 + 8 * half + i;
            int col = h * HD + j * 16 + nn;
            float o = acc[j][i] / lrow[i];
            Ob[((size_t)(b * SEQ + row)) * DIMM + col] = f2bf(o);
        }
    }
}

// ---------- host launcher ----------

extern "C" void kernel_launch(void* const* d_in, const int* in_sizes, int n_in,
                              void* d_out, int out_size, void* d_ws, size_t ws_size,
                              hipStream_t stream) {
    const float*         x     = (const float*)d_in[0];
    const unsigned char* xmask = (const unsigned char*)d_in[1];
    const float*         fcos  = (const float*)d_in[2];
    const float*         fsin  = (const float*)d_in[3];
    const float*         Wqkv  = (const float*)d_in[4];
    const float*         Wout  = (const float*)d_in[5];
    const float*         qw    = (const float*)d_in[6];
    const float*         kw    = (const float*)d_in[7];
    float*               out   = (float*)d_out;

    char* ws = (char*)d_ws;
    size_t off = 0;
    unsigned short* Xb    = (unsigned short*)(ws + off); off += (size_t)NTOK * DIMM * 2;
    unsigned short* Wqkvb = (unsigned short*)(ws + off); off += (size_t)QKVD * DIMM * 2;
    unsigned short* Woutb = (unsigned short*)(ws + off); off += (size_t)DIMM * DIMM * 2;
    float*          qkv   = (float*)(ws + off);          off += (size_t)NTOK * QKVD * 4;
    unsigned short* Qb    = (unsigned short*)(ws + off); off += (size_t)BSZ * NH  * SEQ * HD * 2;
    unsigned short* Kb    = (unsigned short*)(ws + off); off += (size_t)BSZ * NKV * SEQ * HD * 2;
    unsigned short* Vb    = (unsigned short*)(ws + off); off += (size_t)BSZ * NKV * SEQ * HD * 2;
    unsigned short* Ob    = (unsigned short*)(ws + off); off += (size_t)NTOK * DIMM * 2;

    // 1. fp32 -> bf16 conversions
    cvt_bf16_kernel<<<((size_t)NTOK * DIMM) / 1024, 256, 0, stream>>>(x,    Xb,    (long long)NTOK * DIMM);
    cvt_bf16_kernel<<<((size_t)QKVD * DIMM) / 1024, 256, 0, stream>>>(Wqkv, Wqkvb, (long long)QKVD * DIMM);
    cvt_bf16_kernel<<<((size_t)DIMM * DIMM) / 1024, 256, 0, stream>>>(Wout, Woutb, (long long)DIMM * DIMM);

    // 2. QKV projection: qkv[NTOK][QKVD] = Xb @ Wqkvb^T
    dim3 g1(QKVD / 128, NTOK / 64);
    wmma_gemm_nt<<<g1, 128, 0, stream>>>(Xb, Wqkvb, qkv, NTOK, QKVD, DIMM);

    // 3. RMSNorm + RoPE + scatter to per-head Q/K/V (bf16)
    dim3 g2(NTOK, 8);
    rms_rope_scatter<<<g2, 128, 0, stream>>>(qkv, fcos, fsin, qw, kw, Qb, Kb, Vb);

    // 4. GQA flash attention -> Ob[NTOK][DIMM] (bf16)
    dim3 g3(SEQ / 64, NH, BSZ);
    flash_attn<<<g3, 128, 0, stream>>>(Qb, Kb, Vb, xmask, Ob);

    // 5. Output projection: out = Ob @ Woutb^T (fp32 result)
    dim3 g4(DIMM / 128, NTOK / 64);
    wmma_gemm_nt<<<g4, 128, 0, stream>>>(Ob, Woutb, out, NTOK, DIMM, DIMM);
}
